// MainLSTM_43465069035646
// MI455X (gfx1250) — matrix operand order repaired and checked by reading, not compiled, not run on hardware
//
#include <hip/hip_runtime.h>
#include <hip/hip_bf16.h>

// ---------------------------------------------------------------------------
// MainLSTM on MI455X (gfx1250)
//   Phase 1: x_proj = x @ W_xg : bf16 WMMA GEMM, fragments loaded straight
//            from global with b128 loads (A row-major, B pre-transposed).
//   Phase 2: sequential LSTM scan: persistent kernel, 128 WGs, 2 device-wide
//            barriers per step, float4 GEMVs against transposed fp32 weights
//            resident in the 192 MB L2.
// ---------------------------------------------------------------------------

typedef __attribute__((ext_vector_type(16))) __bf16 v16bf;
typedef __attribute__((ext_vector_type(8)))  __bf16 v8bf;
typedef __attribute__((ext_vector_type(8)))  float  v8f;

union FragAB {
  v16bf v;
  v8bf  h[2];
};

#define INP_DIM 512
#define HID_DIM 1024
#define GATE_DIM (4 * HID_DIM)
#define SEQ_LEN 4096

#define REC_BLOCKS 128
#define REC_THREADS 256

// ---------------- setup kernels ----------------

__global__ __launch_bounds__(256) void k_cast_bf16(const float* __restrict__ in,
                                                   __bf16* __restrict__ out, int n) {
  int i = blockIdx.x * blockDim.x + threadIdx.x;
  if (i < n) out[i] = (__bf16)in[i];
}

// in[R][C] fp32 row-major -> out[C][R] bf16
__global__ __launch_bounds__(256) void k_cast_transpose_bf16(const float* __restrict__ in,
                                                             __bf16* __restrict__ out,
                                                             int R, int C) {
  __shared__ float tile[32][33];
  int c0 = blockIdx.x * 32, r0 = blockIdx.y * 32;
  int tx = threadIdx.x & 31, ty = threadIdx.x >> 5;
  #pragma unroll
  for (int i = 0; i < 32; i += 8) {
    int r = r0 + ty + i, c = c0 + tx;
    if (r < R && c < C) tile[ty + i][tx] = in[(long)r * C + c];
  }
  __syncthreads();
  #pragma unroll
  for (int i = 0; i < 32; i += 8) {
    int r = c0 + ty + i, c = r0 + tx;
    if (r < C && c < R) out[(long)r * R + c] = (__bf16)tile[tx][ty + i];
  }
}

// in[R][C] fp32 row-major -> out[C][R] fp32
__global__ __launch_bounds__(256) void k_transpose(const float* __restrict__ in,
                                                   float* __restrict__ out,
                                                   int R, int C) {
  __shared__ float tile[32][33];
  int c0 = blockIdx.x * 32, r0 = blockIdx.y * 32;
  int tx = threadIdx.x & 31, ty = threadIdx.x >> 5;
  #pragma unroll
  for (int i = 0; i < 32; i += 8) {
    int r = r0 + ty + i, c = c0 + tx;
    if (r < R && c < C) tile[ty + i][tx] = in[(long)r * C + c];
  }
  __syncthreads();
  #pragma unroll
  for (int i = 0; i < 32; i += 8) {
    int r = c0 + ty + i, c = r0 + tx;
    if (r < C && c < R) out[(long)r * R + c] = tile[tx][ty + i];
  }
}

// ---------------- phase 1: WMMA GEMM ----------------
// C[M,N] = A[M,K] * B^T   (A bf16 row-major [M,K], BT bf16 [N,K])
// Block: 8 waves, block tile 128(M) x 64(N). Wave tile 16(M) x 64(N):
// per K-step: 1 A-fragment + 4 B-fragments, 4 WMMAs. All loads are b128
// directly from global (L2-resident operands), no LDS.

__global__ __launch_bounds__(256) void k_xproj_wmma(const __bf16* __restrict__ A,
                                                    const __bf16* __restrict__ BT,
                                                    float* __restrict__ C,
                                                    int M, int N, int K) {
  const int bm = blockIdx.y * 128;
  const int bn = blockIdx.x * 64;
  const int wave = threadIdx.x >> 5;
  const int lane = threadIdx.x & 31;
  const int half = lane >> 4;        // lane half-group
  const int ml = lane & 15;

  const int wm = bm + wave * 16;     // this wave's 16 output rows

  // Per-lane base pointers.
  // A fragment lane row = ml; elements: K = kk + 16*half + [0..15]
  const __bf16* arow = A + (long)(wm + ml) * K;
  // B fragment lane col = ml (per 16-col tile); elements K = kk+16*half+[0..15]
  const __bf16* brow0 = BT + (long)(bn + 0  + ml) * K;
  const __bf16* brow1 = BT + (long)(bn + 16 + ml) * K;
  const __bf16* brow2 = BT + (long)(bn + 32 + ml) * K;
  const __bf16* brow3 = BT + (long)(bn + 48 + ml) * K;

  v8f acc0 = {}, acc1 = {}, acc2 = {}, acc3 = {};

  for (int kk = 0; kk < K; kk += 32) {
    FragAB fa, fb0, fb1, fb2, fb3;
    fa.h[0] = *(const v8bf*)(arow + kk + 16 * half);
    fa.h[1] = *(const v8bf*)(arow + kk + 16 * half + 8);

    fb0.h[0] = *(const v8bf*)(brow0 + kk + 16 * half);
    fb0.h[1] = *(const v8bf*)(brow0 + kk + 16 * half + 8);
    fb1.h[0] = *(const v8bf*)(brow1 + kk + 16 * half);
    fb1.h[1] = *(const v8bf*)(brow1 + kk + 16 * half + 8);
    fb2.h[0] = *(const v8bf*)(brow2 + kk + 16 * half);
    fb2.h[1] = *(const v8bf*)(brow2 + kk + 16 * half + 8);
    fb3.h[0] = *(const v8bf*)(brow3 + kk + 16 * half);
    fb3.h[1] = *(const v8bf*)(brow3 + kk + 16 * half + 8);

    acc0 = __builtin_amdgcn_wmma_f32_16x16x32_bf16(false, fa.v, false, fb0.v,
                                                   (short)0, acc0, false, false);
    acc1 = __builtin_amdgcn_wmma_f32_16x16x32_bf16(false, fa.v, false, fb1.v,
                                                   (short)0, acc1, false, false);
    acc2 = __builtin_amdgcn_wmma_f32_16x16x32_bf16(false, fa.v, false, fb2.v,
                                                   (short)0, acc2, false, false);
    acc3 = __builtin_amdgcn_wmma_f32_16x16x32_bf16(false, fa.v, false, fb3.v,
                                                   (short)0, acc3, false, false);
  }

  // C layout: vgpr r, lane -> row = 8*half + r, col = ml (per 16x16 tile)
  #pragma unroll
  for (int r = 0; r < 8; ++r) {
    long row = (long)(wm + 8 * half + r) * N;
    C[row + bn + 0  + ml] = acc0[r];
    C[row + bn + 16 + ml] = acc1[r];
    C[row + bn + 32 + ml] = acc2[r];
    C[row + bn + 48 + ml] = acc3[r];
  }
}

// ---------------- phase 2: persistent recurrence ----------------

__device__ __forceinline__ float sigmoidf_(float x) {
  return 1.0f / (1.0f + __expf(-x));
}

__device__ __forceinline__ void grid_sync(int* bar, int& epoch) {
  __syncthreads();
  epoch += (int)gridDim.x;
  if (threadIdx.x == 0) {
    __hip_atomic_fetch_add(bar, 1, __ATOMIC_ACQ_REL, __HIP_MEMORY_SCOPE_AGENT);
    while (__hip_atomic_load(bar, __ATOMIC_ACQUIRE, __HIP_MEMORY_SCOPE_AGENT) < epoch)
      __builtin_amdgcn_s_sleep(1);
  }
  __syncthreads();
}

__device__ __forceinline__ float wave_reduce(float acc) {
  #pragma unroll
  for (int off = 16; off > 0; off >>= 1)
    acc += __shfl_xor(acc, off, 32);
  return acc;
}

__device__ __forceinline__ float dot4_acc(float4 a, float4 b, float acc) {
  acc = fmaf(a.x, b.x, acc);
  acc = fmaf(a.y, b.y, acc);
  acc = fmaf(a.z, b.z, acc);
  acc = fmaf(a.w, b.w, acc);
  return acc;
}

__global__ __launch_bounds__(REC_THREADS) void k_recurrence(
    const float* __restrict__ xproj,  // [T, 4H]
    const float* __restrict__ WhgT,   // [4H, H]
    const float* __restrict__ WhrT,   // [H, H]
    const float* __restrict__ WcrT,   // [H, H]
    const float* __restrict__ b_g,    // [4H]
    const float* __restrict__ b_r,    // [H]
    float* __restrict__ h,            // [H]   (zeroed)
    float* __restrict__ cbuf0,        // [H]   (zeroed)  ping
    float* __restrict__ cbuf1,        // [H]   (zeroed)  pong
    float* __restrict__ gates,        // [4H]  scratch
    float* __restrict__ rh,           // [H]   scratch (h_prev @ W_hr)
    float* __restrict__ out,          // [T, H]
    int* bar, int T) {
  const int H = HID_DIM, G = GATE_DIM;
  const int lane = threadIdx.x & 31;
  const int wave_g = (blockIdx.x * blockDim.x + threadIdx.x) >> 5;
  const int nwaves = (gridDim.x * blockDim.x) >> 5;
  int epoch = 0;

  for (int t = 0; t < T; ++t) {
    // ---- Phase A: gates = xproj[t] + h@W_hg + b_g ; rh = h@W_hr ----
    float4 hreg[8];
    #pragma unroll
    for (int i = 0; i < 8; ++i)
      hreg[i] = ((const float4*)h)[lane + 32 * i];

    for (int r = wave_g; r < G + H; r += nwaves) {
      const float4* wrow = (r < G) ? (const float4*)&WhgT[(long)r * H]
                                   : (const float4*)&WhrT[(long)(r - G) * H];
      float acc = 0.0f;
      #pragma unroll
      for (int i = 0; i < 8; ++i)
        acc = dot4_acc(hreg[i], wrow[lane + 32 * i], acc);
      acc = wave_reduce(acc);
      if (lane == 0) {
        if (r < G) gates[r] = acc + xproj[(long)t * G + r] + b_g[r];
        else       rh[r - G] = acc;
      }
    }
    grid_sync(bar, epoch);

    // ---- Phase C: each lane recomputes its own c_new slice in registers,
    //      then rc = c_new@W_cr ; h = sig(o)*tanh(rh + rc + b_r) ----
    const float* crd = (t & 1) ? cbuf1 : cbuf0;   // c_{t-1}
    float*       cwr = (t & 1) ? cbuf0 : cbuf1;   // c_t

    float4 creg[8];
    #pragma unroll
    for (int i = 0; i < 8; ++i) {
      int p = lane + 32 * i;
      float4 co = ((const float4*)crd)[p];
      float4 gf = ((const float4*)(gates + H))[p];
      float4 gi = ((const float4*)(gates + 2 * H))[p];
      float4 gg = ((const float4*)(gates + 3 * H))[p];
      float4 cn;
      cn.x = sigmoidf_(gf.x) * co.x + sigmoidf_(gi.x) * tanhf(gg.x);
      cn.y = sigmoidf_(gf.y) * co.y + sigmoidf_(gi.y) * tanhf(gg.y);
      cn.z = sigmoidf_(gf.z) * co.z + sigmoidf_(gi.z) * tanhf(gg.z);
      cn.w = sigmoidf_(gf.w) * co.w + sigmoidf_(gi.w) * tanhf(gg.w);
      creg[i] = cn;
    }
    // one wave persists c_t for the next step (safe: double-buffered)
    if (wave_g == 0) {
      #pragma unroll
      for (int i = 0; i < 8; ++i)
        ((float4*)cwr)[lane + 32 * i] = creg[i];
    }

    for (int j = wave_g; j < H; j += nwaves) {
      const float4* wrow = (const float4*)&WcrT[(long)j * H];
      float acc = 0.0f;
      #pragma unroll
      for (int i = 0; i < 8; ++i)
        acc = dot4_acc(creg[i], wrow[lane + 32 * i], acc);
      acc = wave_reduce(acc);
      if (lane == 0) {
        float o_t = sigmoidf_(gates[j]);
        float hn = o_t * tanhf(rh[j] + acc + b_r[j]);
        out[(long)t * H + j] = hn;
        h[j] = hn;
      }
    }
    grid_sync(bar, epoch);
  }
}

// ---------------- launch ----------------

extern "C" void kernel_launch(void* const* d_in, const int* in_sizes, int n_in,
                              void* d_out, int out_size, void* d_ws, size_t ws_size,
                              hipStream_t stream) {
  const float* x    = (const float*)d_in[0];  // [4096, 512]
  const float* W_xg = (const float*)d_in[1];  // [512, 4096]
  const float* W_hg = (const float*)d_in[2];  // [1024, 4096]
  const float* b_g  = (const float*)d_in[3];  // [4096]
  const float* W_cr = (const float*)d_in[4];  // [1024, 1024]
  const float* W_hr = (const float*)d_in[5];  // [1024, 1024]
  const float* b_r  = (const float*)d_in[6];  // [1024]
  float* out = (float*)d_out;                 // [4096, 1024]

  char* ws = (char*)d_ws;
  // workspace layout (bytes)
  int*    bar    = (int*)   (ws + 0);
  float*  h_st   = (float*) (ws + 4096);
  float*  c_st0  = (float*) (ws + 8192);
  float*  c_st1  = (float*) (ws + 12288);
  float*  rh     = (float*) (ws + 16384);
  float*  gates  = (float*) (ws + 20480);                 // 16 KB
  __bf16* x_bf   = (__bf16*)(ws + 40960);                 // 4 MB
  __bf16* WxgT_bf= (__bf16*)(ws + 40960 + 4194304);       // 4 MB  [N,K] bf16
  float*  xproj  = (float*) (ws + 40960 + 2 * 4194304);   // 64 MB
  float*  WhgT   = (float*) (ws + 40960 + 2 * 4194304 + 67108864);
  float*  WhrT   = (float*) (ws + 40960 + 2 * 4194304 + 67108864 + 16777216);
  float*  WcrT   = (float*) (ws + 40960 + 2 * 4194304 + 67108864 + 16777216 + 4194304);

  // zero barrier counter + h0 + c0/c1
  hipMemsetAsync(d_ws, 0, 16384, stream);

  // x -> bf16 (row-major [T,K])
  {
    int n = SEQ_LEN * INP_DIM;
    k_cast_bf16<<<(n + 255) / 256, 256, 0, stream>>>(x, x_bf, n);
  }
  // W_xg [K,N] fp32 -> [N,K] bf16 (so B fragments are contiguous b128 loads)
  k_cast_transpose_bf16<<<dim3(GATE_DIM / 32, INP_DIM / 32), 256, 0, stream>>>(
      W_xg, WxgT_bf, INP_DIM, GATE_DIM);

  // transposes for coalesced GEMV rows
  k_transpose<<<dim3(GATE_DIM / 32, HID_DIM / 32), 256, 0, stream>>>(W_hg, WhgT, HID_DIM, GATE_DIM);
  k_transpose<<<dim3(HID_DIM / 32, HID_DIM / 32), 256, 0, stream>>>(W_hr, WhrT, HID_DIM, HID_DIM);
  k_transpose<<<dim3(HID_DIM / 32, HID_DIM / 32), 256, 0, stream>>>(W_cr, WcrT, HID_DIM, HID_DIM);

  // x_proj GEMM via WMMA: M=4096, N=4096, K=512
  k_xproj_wmma<<<dim3(GATE_DIM / 64, SEQ_LEN / 128), 256, 0, stream>>>(
      x_bf, WxgT_bf, xproj, SEQ_LEN, GATE_DIM, INP_DIM);

  // persistent recurrence
  k_recurrence<<<REC_BLOCKS, REC_THREADS, 0, stream>>>(
      xproj, WhgT, WhrT, WcrT, b_g, b_r,
      h_st, c_st0, c_st1, gates, rh, out, bar, SEQ_LEN);
}
